// GATModel_70540542869950
// MI455X (gfx1250) — matrix operand order repaired and checked by reading, not compiled, not run on hardware
//
#include <hip/hip_runtime.h>
#include <hip/hip_fp16.h>

#define N_NODES   20000
#define N_EDGES   640000
#define EDGE_DIM  32
#define NODE_DIM  128
#define HID       64
#define AGG_HID   128
#define N_GRAPHS  64
#define E_TOT     (N_EDGES + N_NODES)   // 660000
#define KP_PAD    4                     // LDS row padding (floats) -> conflict-free row reads

typedef __attribute__((ext_vector_type(16))) _Float16 v16h;
typedef __attribute__((ext_vector_type(8)))  float    v8f;
typedef __attribute__((ext_vector_type(4)))  int      v4i;

#if defined(__has_builtin)
#  if __has_builtin(__builtin_amdgcn_global_load_async_to_lds_b128)
#    define HAVE_ASYNC_COPY 1
#  endif
#  if __has_builtin(__builtin_amdgcn_s_wait_asynccnt)
#    define HAVE_WAIT_ASYNC 1
#  endif
#endif

#define AS_GLOBAL __attribute__((address_space(1)))
#define AS_LOCAL  __attribute__((address_space(3)))

// ---------- helpers ----------
__device__ __forceinline__ unsigned encf(float f) {
    unsigned u = __float_as_uint(f);
    return (u & 0x80000000u) ? ~u : (u | 0x80000000u);
}
__device__ __forceinline__ float decf(unsigned u) {
    unsigned b = (u & 0x80000000u) ? (u ^ 0x80000000u) : ~u;
    return __uint_as_float(b);
}

__global__ void fill_f32(float* p, float v, size_t n) {
    size_t i = (size_t)blockIdx.x * blockDim.x + threadIdx.x;
    if (i < n) p[i] = v;
}
__global__ void fill_u32(unsigned* p, unsigned v, size_t n) {
    size_t i = (size_t)blockIdx.x * blockDim.x + threadIdx.x;
    if (i < n) p[i] = v;
}

// ---------- self-loop edge attr: mean of incoming edge_attr ----------
__global__ void deg_loop_kernel(const int* __restrict__ dst, const float* __restrict__ eattr,
                                float* __restrict__ deg, float* __restrict__ loopAttr) {
    size_t tid = (size_t)blockIdx.x * blockDim.x + threadIdx.x;
    size_t e = tid >> 5; int k = (int)(tid & 31);
    if (e >= N_EDGES) return;
    int d = dst[e];
    atomicAdd(&loopAttr[(size_t)d * EDGE_DIM + k], eattr[e * EDGE_DIM + k]);
    if (k == 0) atomicAdd(&deg[d], 1.0f);
}
__global__ void loop_norm_kernel(float* __restrict__ loopAttr, const float* __restrict__ deg) {
    size_t tid = (size_t)blockIdx.x * blockDim.x + threadIdx.x;
    if (tid >= (size_t)N_NODES * EDGE_DIM) return;
    int n = (int)(tid / EDGE_DIM);
    loopAttr[tid] /= fmaxf(deg[n], 1.0f);
}

// ---------- weight pre-pack: W[K,M] f32 -> f16 in WMMA B-register order ----------
// Packed element (kt, nt, lane, e) = W[kt*32 + (e&7) + 16*(e>>3) + 8*(lane>>4)][nt*16 + (lane&15)]
// so a B tile is one contiguous v16h (32 B) load per lane.
__global__ void pack_w_f16(const float* __restrict__ W, _Float16* __restrict__ wp,
                           int K, int M) {
    size_t idx = (size_t)blockIdx.x * blockDim.x + threadIdx.x;
    size_t total = (size_t)(K >> 5) * (M >> 4) * 512;
    if (idx >= total) return;
    int e    = (int)(idx & 15);
    int lane = (int)((idx >> 4) & 31);
    size_t t = idx >> 9;                       // kt*numNT + nt
    int numNT = M >> 4;
    int nt = (int)(t % numNT), kt = (int)(t / numNT);
    int k   = kt * 32 + (e & 7) + ((e >> 3) << 4) + ((lane >> 4) << 3);
    int col = nt * 16 + (lane & 15);
    wp[idx] = (_Float16)W[(size_t)k * M + col];
}

// ---------- WMMA GEMM: out[N,M] = A[N,K] @ W[K,M] + bias (opt relu) ----------
// Block = 4 waves sharing one 16-row A panel staged in LDS (async copy when
// available); each wave owns one 16-col tile. N%16==0, K%32==0, M%64==0, K<=128.
__global__ void wmma_gemm(const float* __restrict__ A, const _Float16* __restrict__ Wp,
                          const float* __restrict__ bias, float* __restrict__ out,
                          int N, int K, int M, int relu) {
    __shared__ float ldsA[16 * (128 + KP_PAD)];
    const int KP = K + KP_PAD;
    int tid = threadIdx.y * 32 + threadIdx.x;
    int tileM = blockIdx.x;

    // ---- stage A panel (16 rows x K f32) into LDS, 16B chunks ----
    int chunksPerRow = K >> 2;
    int totalChunks = chunksPerRow << 4;               // multiple of 128
    for (int cid = tid; cid < totalChunks; cid += 128) {
        int row = cid / chunksPerRow, cc = cid - row * chunksPerRow;
        const float* gsrc = A + (size_t)(tileM * 16 + row) * K + (cc << 2);
        float* ldst = &ldsA[row * KP + (cc << 2)];
#if defined(HAVE_ASYNC_COPY)
        __builtin_amdgcn_global_load_async_to_lds_b128(
            (AS_GLOBAL v4i*)(void*)gsrc,
            (AS_LOCAL  v4i*)(void*)ldst, 0, 0);
#else
        *(float4*)ldst = *(const float4*)gsrc;
#endif
    }
#if defined(HAVE_ASYNC_COPY)
#  if defined(HAVE_WAIT_ASYNC)
    __builtin_amdgcn_s_wait_asynccnt(0);
#  else
    asm volatile("s_wait_asynccnt 0x0" ::: "memory");
#  endif
#endif
    __syncthreads();

    // ---- per-wave 16x16 tile ----
    int lane = threadIdx.x, hs = lane >> 4, l15 = lane & 15;
    int tileN = blockIdx.y * 4 + threadIdx.y;
    int numNT = M >> 4;
    v8f c = {};
    int kt = 0;
    for (int k0 = 0; k0 < K; k0 += 32, ++kt) {
        v16h a, b;
        const float* arow = &ldsA[l15 * KP + k0 + (hs << 3)];
        #pragma unroll
        for (int e = 0; e < 8; ++e) {          // A: K = (e&7) + ((e>>3)<<4) + (hs<<3)
            a[e]     = (_Float16)arow[e];
            a[e + 8] = (_Float16)arow[e + 16];
        }
        b = *(const v16h*)(Wp + (((size_t)kt * numNT + tileN) * 32 + lane) * 16);
        c = __builtin_amdgcn_wmma_f32_16x16x32_f16(false, a, false, b, (short)0, c, false, false);
    }
    int col = tileN * 16 + l15;
    float bv = bias ? bias[col] : 0.0f;
    #pragma unroll
    for (int r = 0; r < 8; ++r) {              // D: row = r + 8*hs, col = l15
        int row = tileM * 16 + r + (hs << 3);
        float v = c[r] + bv;
        if (relu) v = fmaxf(v, 0.0f);
        out[(size_t)row * M + col] = v;
    }
}

// ---------- fused edge kernel: e = eattr@We (WMMA), logits + segment max ----------
// One wave handles 16 edges. K = EDGE_DIM = 32 -> single WMMA per 16-col tile.
// We is pre-packed f16 (kt = 0 only).
__global__ void edge_logit_wmma(const int* __restrict__ src, const int* __restrict__ dst,
                                const float* __restrict__ eattr, const float* __restrict__ loopAttr,
                                const _Float16* __restrict__ Wep, const float* __restrict__ att,
                                const float* __restrict__ xl, const float* __restrict__ xr,
                                float* __restrict__ logit, unsigned* __restrict__ lmaxEnc,
                                int H, int C, int M) {
    int lane = threadIdx.x;
    int hs = lane >> 4, l15 = lane & 15;
    int ebase = blockIdx.x * 16;

    // A tile: 16 edges x 32 attrs (self-loop rows read loopAttr)
    int arowIdx = ebase + l15;
    const float* arow = (arowIdx < N_EDGES) ? (eattr + (size_t)arowIdx * EDGE_DIM)
                                            : (loopAttr + (size_t)(arowIdx - N_EDGES) * EDGE_DIM);
    v16h a;
    #pragma unroll
    for (int e = 0; e < 8; ++e) {
        a[e]     = (_Float16)arow[e + (hs << 3)];
        a[e + 8] = (_Float16)arow[e + 16 + (hs << 3)];
    }
    // edge endpoints for this lane's 8 D-rows (row = r + 8*hs)
    int s_[8], d_[8];
    #pragma unroll
    for (int r = 0; r < 8; ++r) {
        int row = ebase + r + (hs << 3);
        if (row < N_EDGES) { s_[r] = src[row]; d_[r] = dst[row]; }
        else               { s_[r] = d_[r] = row - N_EDGES; }
    }
    int tilesPerH = C / 16;
    for (int h = 0; h < H; ++h) {
        float acc[8];
        #pragma unroll
        for (int r = 0; r < 8; ++r) acc[r] = 0.0f;
        for (int t = 0; t < tilesPerH; ++t) {
            int colbase = h * C + t * 16;
            v16h b = *(const v16h*)(Wep + ((size_t)(colbase >> 4) * 32 + lane) * 16);
            v8f c = {};
            c = __builtin_amdgcn_wmma_f32_16x16x32_f16(false, a, false, b, (short)0, c, false, false);
            int col = colbase + l15;
            float av = att[col];
            #pragma unroll
            for (int r = 0; r < 8; ++r) {
                float m = c[r] + xl[(size_t)s_[r] * M + col] + xr[(size_t)d_[r] * M + col];
                m = (m > 0.0f) ? m : 0.2f * m;          // LeakyReLU(0.2)
                acc[r] += m * av;
            }
        }
        // reduce across the 16 lanes holding the same rows (xor masks keep bit4 fixed)
        #pragma unroll
        for (int r = 0; r < 8; ++r) {
            float v = acc[r];
            for (int msk = 1; msk < 16; msk <<= 1) v += __shfl_xor(v, msk, 32);
            acc[r] = v;
        }
        if (l15 == 0) {
            #pragma unroll
            for (int r = 0; r < 8; ++r) {
                int row = ebase + r + (hs << 3);
                logit[(size_t)row * H + h] = acc[r];
                atomicMax(&lmaxEnc[(size_t)d_[r] * H + h], encf(acc[r]));
            }
        }
    }
}

// ---------- softmax numerator + segment sum ----------
__global__ void edge_softmax(const int* __restrict__ dst, const float* __restrict__ logit,
                             const unsigned* __restrict__ lmaxEnc,
                             float* __restrict__ alpha, float* __restrict__ asum, int H) {
    size_t tid = (size_t)blockIdx.x * blockDim.x + threadIdx.x;
    if (tid >= (size_t)E_TOT * H) return;
    size_t e = tid / H; int h = (int)(tid % H);
    int d = (e < N_EDGES) ? dst[e] : (int)(e - N_EDGES);
    float a = expf(logit[tid] - decf(lmaxEnc[(size_t)d * H + h]));
    alpha[tid] = a;
    atomicAdd(&asum[(size_t)d * H + h], a);
}

// ---------- scatter-sum aggregation (block per edge, thread per channel) ----------
__global__ void edge_aggregate(const int* __restrict__ src, const int* __restrict__ dst,
                               const float* __restrict__ xl, const float* __restrict__ alpha,
                               float* __restrict__ acc, int H, int C, int M) {
    int e = blockIdx.x; int col = threadIdx.x;
    int s, d;
    if (e < N_EDGES) { s = src[e]; d = dst[e]; } else { s = d = e - N_EDGES; }
    int h = col / C;
    float a = alpha[(size_t)e * H + h];
    atomicAdd(&acc[(size_t)d * M + col], a * xl[(size_t)s * M + col]);
}

// ---------- per-node normalize + output bias (in place) ----------
__global__ void node_finalize(float* __restrict__ acc, const float* __restrict__ asum,
                              const float* __restrict__ b_out, int H, int C, int M) {
    size_t tid = (size_t)blockIdx.x * blockDim.x + threadIdx.x;
    if (tid >= (size_t)N_NODES * M) return;
    size_t n = tid / M; int col = (int)(tid % M); int h = col / C;
    acc[tid] = acc[tid] / (asum[n * H + h] + 1e-16f) + b_out[col];
}

// ---------- BatchNorm stats (block per channel) ----------
__global__ void bn_stats(const float* __restrict__ x, float* __restrict__ mean,
                         float* __restrict__ rstd, int M) {
    int ch = blockIdx.x;
    __shared__ float s1[256], s2[256];
    float a = 0.0f, b = 0.0f;
    for (int n = threadIdx.x; n < N_NODES; n += blockDim.x) {
        float v = x[(size_t)n * M + ch];
        a += v; b += v * v;
    }
    s1[threadIdx.x] = a; s2[threadIdx.x] = b;
    __syncthreads();
    for (int s = 128; s > 0; s >>= 1) {
        if (threadIdx.x < s) { s1[threadIdx.x] += s1[threadIdx.x + s]; s2[threadIdx.x] += s2[threadIdx.x + s]; }
        __syncthreads();
    }
    if (threadIdx.x == 0) {
        float mu = s1[0] / (float)N_NODES;
        float var = s2[0] / (float)N_NODES - mu * mu;   // biased var (torch BN)
        mean[ch] = mu;
        rstd[ch] = rsqrtf(var + 1e-5f);
    }
}
__global__ void bn_apply(float* __restrict__ x, const float* __restrict__ mean,
                         const float* __restrict__ rstd, const float* __restrict__ gamma,
                         const float* __restrict__ beta, int M, int relu) {
    size_t tid = (size_t)blockIdx.x * blockDim.x + threadIdx.x;
    if (tid >= (size_t)N_NODES * M) return;
    int ch = (int)(tid % M);
    float v = gamma[ch] * (x[tid] - mean[ch]) * rstd[ch] + beta[ch];
    if (relu) v = fmaxf(v, 0.0f);
    x[tid] = v;
}

// ---------- row dot (one wave per row): out[n] = A[n,:]@w + b ----------
__global__ void rowdot(const float* __restrict__ A, const float* __restrict__ w,
                       const float* __restrict__ bptr, float* __restrict__ out, int K) {
    int n = blockIdx.x, lane = threadIdx.x;
    float s = 0.0f;
    for (int k = lane; k < K; k += 32) s += A[(size_t)n * K + k] * w[k];
    for (int m = 16; m; m >>= 1) s += __shfl_xor(s, m, 32);
    if (lane == 0) out[n] = s + bptr[0];
}

// ---------- attention pooling ----------
__global__ void pool_max(const float* __restrict__ g, const int* __restrict__ batch,
                         unsigned* __restrict__ gmax) {
    int n = blockIdx.x * blockDim.x + threadIdx.x;
    if (n >= N_NODES) return;
    atomicMax(&gmax[batch[n]], encf(g[n]));
}
__global__ void pool_w(const float* __restrict__ g, const int* __restrict__ batch,
                       const unsigned* __restrict__ gmax, float* __restrict__ wexp,
                       float* __restrict__ wsum) {
    int n = blockIdx.x * blockDim.x + threadIdx.x;
    if (n >= N_NODES) return;
    float w = expf(g[n] - decf(gmax[batch[n]]));
    wexp[n] = w;
    atomicAdd(&wsum[batch[n]], w);
}
__global__ void pool_acc(const float* __restrict__ h, const float* __restrict__ wexp,
                         const int* __restrict__ batch, float* __restrict__ pooled, int M) {
    size_t tid = (size_t)blockIdx.x * blockDim.x + threadIdx.x;
    if (tid >= (size_t)N_NODES * M) return;
    size_t n = tid / M; int ch = (int)(tid % M);
    atomicAdd(&pooled[(size_t)batch[n] * M + ch], wexp[n] * h[tid]);
}
__global__ void final_reg(const float* __restrict__ pooled, const float* __restrict__ wsum,
                          const float* __restrict__ regW, const float* __restrict__ regb,
                          float* __restrict__ out, int M) {
    int b = blockIdx.x, lane = threadIdx.x;
    float s = 0.0f;
    for (int k = lane; k < M; k += 32) s += pooled[(size_t)b * M + k] * regW[k];
    for (int m = 16; m; m >>= 1) s += __shfl_xor(s, m, 32);
    if (lane == 0) out[b] = s / (wsum[b] + 1e-16f) + regb[0];
}

// =====================================================================
extern "C" void kernel_launch(void* const* d_in, const int* in_sizes, int n_in,
                              void* d_out, int out_size, void* d_ws, size_t ws_size,
                              hipStream_t stream) {
    (void)in_sizes; (void)n_in; (void)out_size; (void)ws_size;
    const float* x     = (const float*)d_in[0];
    const int*   ei    = (const int*)d_in[1];
    const int*   src   = ei;
    const int*   dst   = ei + N_EDGES;
    const float* eattr = (const float*)d_in[2];
    const int*   batch = (const int*)d_in[3];

    struct ConvP { const float *Wl,*bl,*Wr,*br,*We,*att,*b_out,*gamma,*beta; };
    ConvP cp[3];
    int base = 4;
    for (int i = 0; i < 3; ++i) {
        cp[i].Wl    = (const float*)d_in[base + 0];
        cp[i].bl    = (const float*)d_in[base + 1];
        cp[i].Wr    = (const float*)d_in[base + 2];
        cp[i].br    = (const float*)d_in[base + 3];
        cp[i].We    = (const float*)d_in[base + 4];
        cp[i].att   = (const float*)d_in[base + 5];
        cp[i].b_out = (const float*)d_in[base + 6];
        cp[i].gamma = (const float*)d_in[base + 7];
        cp[i].beta  = (const float*)d_in[base + 8];
        base += 9;
    }
    const float* gW1  = (const float*)d_in[base + 0];
    const float* gb1  = (const float*)d_in[base + 1];
    const float* gW2  = (const float*)d_in[base + 2];
    const float* gb2  = (const float*)d_in[base + 3];
    const float* regW = (const float*)d_in[base + 4];
    const float* regb = (const float*)d_in[base + 5];
    float* out = (float*)d_out;

    // workspace carve-up (float units, 32B-aligned chunks)
    float* fw = (float*)d_ws;
    size_t o = 0;
    auto alloc = [&](size_t n) { float* p = fw + o; o += (n + 7) & ~(size_t)7; return p; };
    float*    deg      = alloc(N_NODES);
    float*    loopAttr = alloc((size_t)N_NODES * EDGE_DIM);
    float*    xl       = alloc((size_t)N_NODES * 128);
    float*    xr       = alloc((size_t)N_NODES * 128);
    float*    bufA     = alloc((size_t)N_NODES * 128);
    float*    bufB     = alloc((size_t)N_NODES * 128);
    float*    logit    = alloc((size_t)E_TOT * 2);
    float*    alpha    = alloc((size_t)E_TOT * 2);
    float*    gact     = alloc((size_t)N_NODES * AGG_HID);
    float*    gscore   = alloc(N_NODES);
    float*    wexp     = alloc(N_NODES);
    float*    asum     = alloc((size_t)N_NODES * 2);
    float*    bnmean   = alloc(128);
    float*    bnrstd   = alloc(128);
    float*    wsum     = alloc(N_GRAPHS);
    float*    pooled   = alloc((size_t)N_GRAPHS * HID);
    unsigned* lmaxEnc  = (unsigned*)alloc((size_t)N_NODES * 2);
    unsigned* gmaxEnc  = (unsigned*)alloc(N_GRAPHS);
    _Float16* wpack    = (_Float16*)alloc(128 * 128 / 2);  // 16384 halves max

    auto fillf = [&](float* p, float v, size_t n) {
        fill_f32<<<dim3((unsigned)((n + 255) / 256)), 256, 0, stream>>>(p, v, n);
    };
    auto fillu = [&](unsigned* p, unsigned v, size_t n) {
        fill_u32<<<dim3((unsigned)((n + 255) / 256)), 256, 0, stream>>>(p, v, n);
    };
    auto packw = [&](const float* W, int K, int M) {
        size_t total = (size_t)(K / 32) * (M / 16) * 512;
        pack_w_f16<<<dim3((unsigned)((total + 255) / 256)), 256, 0, stream>>>(W, wpack, K, M);
    };

    // 1) self-loop attrs
    fillf(deg, 0.0f, N_NODES);
    fillf(loopAttr, 0.0f, (size_t)N_NODES * EDGE_DIM);
    {
        size_t nt = (size_t)N_EDGES * 32;
        deg_loop_kernel<<<dim3((unsigned)((nt + 255) / 256)), 256, 0, stream>>>(dst, eattr, deg, loopAttr);
        size_t nn = (size_t)N_NODES * EDGE_DIM;
        loop_norm_kernel<<<dim3((unsigned)((nn + 255) / 256)), 256, 0, stream>>>(loopAttr, deg);
    }

    // 2) three GATv2 convs
    const float* convIn[3]  = { x, bufA, bufB };
    float*       convAcc[3] = { bufA, bufB, bufA };
    const int H_[3] = { 2, 2, 1 };
    const int M_[3] = { 128, 128, 64 };
    const int relu_[3] = { 1, 1, 0 };
    const int DIN = 128;
    const int C = HID;

    for (int i = 0; i < 3; ++i) {
        int H = H_[i], M = M_[i];
        dim3 gblk(32, 4);
        dim3 ggrid(N_NODES / 16, M / 64);
        // node linear transforms via WMMA (pre-packed f16 weights)
        packw(cp[i].Wl, DIN, M);
        wmma_gemm<<<ggrid, gblk, 0, stream>>>(convIn[i], wpack, cp[i].bl, xl, N_NODES, DIN, M, 0);
        packw(cp[i].Wr, DIN, M);
        wmma_gemm<<<ggrid, gblk, 0, stream>>>(convIn[i], wpack, cp[i].br, xr, N_NODES, DIN, M, 0);
        // zero accumulators
        fillf(convAcc[i], 0.0f, (size_t)N_NODES * M);
        fillf(asum, 0.0f, (size_t)N_NODES * H);
        fillu(lmaxEnc, 0u, (size_t)N_NODES * H);   // enc(-inf)
        // fused edge matmul + attention logits + segment max
        packw(cp[i].We, EDGE_DIM, M);
        edge_logit_wmma<<<E_TOT / 16, 32, 0, stream>>>(src, dst, eattr, loopAttr, wpack,
                                                       cp[i].att, xl, xr, logit, lmaxEnc, H, C, M);
        // softmax numerator + denominator
        {
            size_t nt = (size_t)E_TOT * H;
            edge_softmax<<<dim3((unsigned)((nt + 255) / 256)), 256, 0, stream>>>(dst, logit, lmaxEnc,
                                                                                 alpha, asum, H);
        }
        // scatter aggregation
        edge_aggregate<<<E_TOT, M, 0, stream>>>(src, dst, xl, alpha, convAcc[i], H, C, M);
        // normalize + bias
        {
            size_t nt = (size_t)N_NODES * M;
            node_finalize<<<dim3((unsigned)((nt + 255) / 256)), 256, 0, stream>>>(convAcc[i], asum,
                                                                                  cp[i].b_out, H, C, M);
        }
        // batchnorm (+relu)
        bn_stats<<<M, 256, 0, stream>>>(convAcc[i], bnmean, bnrstd, M);
        {
            size_t nt = (size_t)N_NODES * M;
            bn_apply<<<dim3((unsigned)((nt + 255) / 256)), 256, 0, stream>>>(convAcc[i], bnmean, bnrstd,
                                                                             cp[i].gamma, cp[i].beta, M,
                                                                             relu_[i]);
        }
    }
    float* hfin = bufA;   // [N_NODES, HID]

    // 3) gate MLP: gact = relu(h@W1+b1) via WMMA, g = gact@W2 + b2
    {
        dim3 gblk(32, 4);
        dim3 ggrid(N_NODES / 16, AGG_HID / 64);
        packw(gW1, HID, AGG_HID);
        wmma_gemm<<<ggrid, gblk, 0, stream>>>(hfin, wpack, gb1, gact, N_NODES, HID, AGG_HID, 1);
        rowdot<<<N_NODES, 32, 0, stream>>>(gact, gW2, gb2, gscore, AGG_HID);
    }

    // 4) attentional pooling + regression
    fillu(gmaxEnc, 0u, N_GRAPHS);
    fillf(wsum, 0.0f, N_GRAPHS);
    fillf(pooled, 0.0f, (size_t)N_GRAPHS * HID);
    pool_max<<<(N_NODES + 255) / 256, 256, 0, stream>>>(gscore, batch, gmaxEnc);
    pool_w<<<(N_NODES + 255) / 256, 256, 0, stream>>>(gscore, batch, gmaxEnc, wexp, wsum);
    {
        size_t nt = (size_t)N_NODES * HID;
        pool_acc<<<dim3((unsigned)((nt + 255) / 256)), 256, 0, stream>>>(hfin, wexp, batch, pooled, HID);
    }
    final_reg<<<N_GRAPHS, 32, 0, stream>>>(pooled, wsum, regW, regb, out, HID);
}
